// Model_39410619908782
// MI455X (gfx1250) — compile-verified, matrix operand order checked by reading, not run on hardware
//
#include <hip/hip_runtime.h>

// ---------------- problem constants ----------------
#define BQ    8
#define SEQQ  96
#define NVARQ 862
#define LQ    866              // NVAR + NMARK tokens
#define DMQ   512
#define RROWS (BQ*LQ)          // 6928 = 433*16
#define MTILES (RROWS/16)      // 433

typedef __attribute__((ext_vector_type(16))) __bf16 v16bf;
typedef __attribute__((ext_vector_type(8)))  __bf16 v8bf;
typedef __attribute__((ext_vector_type(8)))  float  v8f;

// ---------------- bf16 helpers ----------------
__device__ __forceinline__ unsigned short f2bfu(float f){
  unsigned int u = __builtin_bit_cast(unsigned int, f);
  u += 0x7FFFu + ((u >> 16) & 1u);          // round-to-nearest-even
  return (unsigned short)(u >> 16);
}
__device__ __forceinline__ __bf16 tobf(float f){
  unsigned short u = f2bfu(f);
  return __builtin_bit_cast(__bf16, u);
}

// Fragment from bf16 memory: per lane the 16 values are two contiguous
// 16-byte runs (K = half*8..+7 and 16+half*8..+7)  ->  2x global_load_b128.
__device__ __forceinline__ v16bf load_frag_bf(const __bf16* __restrict__ p, int half){
  union { v16bf v; v8bf h[2]; } f;
  f.h[0] = *(const v8bf*)(p + half*8);
  f.h[1] = *(const v8bf*)(p + 16 + half*8);
  return f.v;
}

enum { EPI_NONE = 0, EPI_BIAS = 1, EPI_RELU = 2, EPI_SOFTPLUS = 3 };

// One wave computes a 16 x (16*NT) tile of A(MxK,bf16,lda) * W(NxK,bf16,ld=K)^T.
// A fragment is reused across the NT sub-tiles each k-step.
template<int EPI, int NT>
__global__ void __launch_bounds__(32)
gemm_wmma(const __bf16* __restrict__ A, int lda,
          const __bf16* __restrict__ W, int K,
          const float* __restrict__ bias,
          float* __restrict__ C, int ldc,
          __bf16* __restrict__ Cbf, int ldbf, int nbf)
{
  const int lane = threadIdx.x;
  const int half = lane >> 4, idx = lane & 15;
  v8f acc[NT];
#pragma unroll
  for(int t = 0; t < NT; ++t) acc[t] = (v8f){};
  const __bf16* ap = A + (size_t)(blockIdx.x*16 + idx)*lda;
  for(int k0 = 0; k0 < K; k0 += 32){
    v16bf a = load_frag_bf(ap + k0, half);
#pragma unroll
    for(int t = 0; t < NT; ++t){
      const __bf16* wp = W + (size_t)((blockIdx.y*NT + t)*16 + idx)*K + k0;
      v16bf b = load_frag_bf(wp, half);
      acc[t] = __builtin_amdgcn_wmma_f32_16x16x32_bf16(false, a, false, b,
                                                       (short)0, acc[t], false, false);
    }
  }
#pragma unroll
  for(int t = 0; t < NT; ++t){
    int col = (blockIdx.y*NT + t)*16 + idx;
    float bv = (EPI != EPI_NONE) ? bias[col] : 0.f;
#pragma unroll
    for(int v = 0; v < 8; ++v){
      int row = blockIdx.x*16 + v + 8*half;     // C/D layout: VGPR v -> M = v + 8*half
      float val = acc[t][v] + bv;
      if(EPI == EPI_RELU)     val = fmaxf(val, 0.f);
      if(EPI == EPI_SOFTPLUS) val = fmaxf(val, 0.f) + log1pf(__expf(-fabsf(val)));
      if(C)                 C[(size_t)row*ldc + col]   = val;
      if(Cbf && col < nbf)  Cbf[(size_t)row*ldbf + col] = tobf(val);
    }
  }
}

// out_bf = bf16( encF + sigmoid(raw @ gw^T + gb) * raw ),  N = K = 512, NT = 4
__global__ void __launch_bounds__(32)
gemm_gate(const __bf16* __restrict__ rawbf, const float* __restrict__ raw,
          const float* __restrict__ encF,
          const __bf16* __restrict__ gw, const float* __restrict__ gb,
          __bf16* __restrict__ outbf)
{
  const int lane = threadIdx.x;
  const int half = lane >> 4, idx = lane & 15;
  v8f acc[4];
#pragma unroll
  for(int t = 0; t < 4; ++t) acc[t] = (v8f){};
  const __bf16* ap = rawbf + (size_t)(blockIdx.x*16 + idx)*DMQ;
  for(int k0 = 0; k0 < DMQ; k0 += 32){
    v16bf a = load_frag_bf(ap + k0, half);
#pragma unroll
    for(int t = 0; t < 4; ++t){
      const __bf16* wp = gw + (size_t)((blockIdx.y*4 + t)*16 + idx)*DMQ + k0;
      v16bf b = load_frag_bf(wp, half);
      acc[t] = __builtin_amdgcn_wmma_f32_16x16x32_bf16(false, a, false, b,
                                                       (short)0, acc[t], false, false);
    }
  }
#pragma unroll
  for(int t = 0; t < 4; ++t){
    int col = (blockIdx.y*4 + t)*16 + idx;
    float bv = gb[col];
#pragma unroll
    for(int v = 0; v < 8; ++v){
      int row = blockIdx.x*16 + v + 8*half;
      float g = 1.f/(1.f + __expf(-(acc[t][v] + bv)));
      size_t o = (size_t)row*DMQ + col;
      outbf[o] = tobf(encF[o] + g*raw[o]);
    }
  }
}

// fused projection: transpose to (B, PRED, NVAR), slice var<862, de-normalize. N=96, NT=2
__global__ void __launch_bounds__(32)
gemm_proj(const __bf16* __restrict__ A, const __bf16* __restrict__ pw,
          const float* __restrict__ pb, const float* __restrict__ meanp,
          const float* __restrict__ stdp, float* __restrict__ out)
{
  const int lane = threadIdx.x;
  const int half = lane >> 4, idx = lane & 15;
  v8f acc[2];
#pragma unroll
  for(int t = 0; t < 2; ++t) acc[t] = (v8f){};
  const __bf16* ap = A + (size_t)(blockIdx.x*16 + idx)*DMQ;
  for(int k0 = 0; k0 < DMQ; k0 += 32){
    v16bf a = load_frag_bf(ap + k0, half);
#pragma unroll
    for(int t = 0; t < 2; ++t){
      const __bf16* wp = pw + (size_t)((blockIdx.y*2 + t)*16 + idx)*DMQ + k0;
      v16bf b = load_frag_bf(wp, half);
      acc[t] = __builtin_amdgcn_wmma_f32_16x16x32_bf16(false, a, false, b,
                                                       (short)0, acc[t], false, false);
    }
  }
#pragma unroll
  for(int t = 0; t < 2; ++t){
    int p = (blockIdx.y*2 + t)*16 + idx;       // 0..95
#pragma unroll
    for(int v = 0; v < 8; ++v){
      int row = blockIdx.x*16 + v + 8*half;
      int b = row / LQ, var = row % LQ;
      if(var < NVARQ){
        float val = acc[t][v] + pb[p];
        out[((size_t)b*SEQQ + p)*NVARQ + var] =
            val*stdp[b*NVARQ+var] + meanp[b*NVARQ+var];
      }
    }
  }
}

// ---------------- f32 -> bf16 weight conversion ----------------
__global__ void cvt_bf16(const float* __restrict__ in, __bf16* __restrict__ out, int n){
  int i = blockIdx.x*blockDim.x + threadIdx.x;
  if(i < n) out[i] = tobf(in[i]);
}

// ---------------- per-series normalization + bf16 token build ----------------
__global__ void norm_tok(const float* __restrict__ x_enc, const float* __restrict__ x_mark,
                         __bf16* __restrict__ tok, float* __restrict__ meanp,
                         float* __restrict__ stdp)
{
  int i = blockIdx.x*blockDim.x + threadIdx.x;
  if(i >= RROWS) return;
  int b = i / LQ, v = i % LQ;
  if(v < NVARQ){
    float m = 0.f;
    for(int s = 0; s < SEQQ; ++s) m += x_enc[((size_t)b*SEQQ+s)*NVARQ + v];
    m *= (1.f/SEQQ);
    float va = 0.f;
    for(int s = 0; s < SEQQ; ++s){
      float d = x_enc[((size_t)b*SEQQ+s)*NVARQ + v] - m; va += d*d;
    }
    float sd = sqrtf(va*(1.f/SEQQ) + 1e-5f);
    meanp[b*NVARQ+v] = m; stdp[b*NVARQ+v] = sd;
    float inv = 1.f/sd;
    for(int s = 0; s < SEQQ; ++s)
      tok[(size_t)i*SEQQ + s] = tobf((x_enc[((size_t)b*SEQQ+s)*NVARQ + v] - m)*inv);
  } else {
    int mk = v - NVARQ;
    for(int s = 0; s < SEQQ; ++s)
      tok[(size_t)i*SEQQ + s] = tobf(x_mark[((size_t)b*SEQQ+s)*4 + mk]);
  }
}

// ---------------- depthwise conv (k=2) + SiLU; xi lives in xz[:,0:512] ----------------
__global__ void conv_silu(const float* __restrict__ xz, const float* __restrict__ cw,
                          const float* __restrict__ cb, float* __restrict__ xc,
                          __bf16* __restrict__ xcbf)
{
  size_t i = (size_t)blockIdx.x*blockDim.x + threadIdx.x;
  if(i >= (size_t)RROWS*DMQ) return;
  int d = (int)(i & (DMQ-1));
  size_t r = i >> 9;
  int l = (int)(r % LQ);
  float cur  = xz[r*1024 + d];
  float prev = (l == 0) ? 0.f : xz[(r-1)*1024 + d];
  float v = prev*cw[d*2+0] + cur*cw[d*2+1] + cb[d];
  float sv = v * (1.f/(1.f + __expf(-v)));
  xc[i]   = sv;
  xcbf[i] = tobf(sv);
}

__global__ void reverse_bf(const unsigned short* __restrict__ in,
                           unsigned short* __restrict__ out)
{
  size_t i = (size_t)blockIdx.x*blockDim.x + threadIdx.x;
  if(i >= (size_t)RROWS*DMQ) return;
  int d = (int)(i & (DMQ-1));
  size_t r = i >> 9;
  int b = (int)(r / LQ), l = (int)(r % LQ);
  out[((size_t)b*LQ + (LQ-1-l))*DMQ + d] = in[i];
}

// ---------------- SSM scan: 16 lanes own the 16 states of a channel ----------------
// block = 256 threads = 16 channel-groups x 16 states; each group carries 16 channels.
// ybf = bf16( (sum_s h*C + D*xc) * silu(z) )
__global__ void __launch_bounds__(256)
mamba_scan(const float* __restrict__ delta, const float* __restrict__ xc,
           const float* __restrict__ dblp, const float* __restrict__ xz,
           const float* __restrict__ alog, const float* __restrict__ dpar,
           __bf16* __restrict__ ybf)
{
  __shared__ float sD[256], sX[256], sBC[32];
  int b = blockIdx.x;
  int dbase = blockIdx.y*256;
  int t = threadIdx.x;
  int s = t & 15, dg = t >> 4;
  float h[16], An[16];
#pragma unroll
  for(int j = 0; j < 16; ++j){
    h[j] = 0.f;
    int d = dbase + dg*16 + j;
    An[j] = -__expf(alog[d*16 + s]);      // A = -exp(A_log)
  }
  for(int l = 0; l < LQ; ++l){
    size_t r = (size_t)b*LQ + l;
    __syncthreads();
    sD[t] = delta[r*DMQ + dbase + t];
    sX[t] = xc[r*DMQ + dbase + t];
    if(t < 32) sBC[t] = dblp[r*64 + 32 + t];  // B (16) then C (16)
    __syncthreads();
    float Bv = sBC[s], Cv = sBC[16+s];
#pragma unroll
    for(int j = 0; j < 16; ++j){
      int dl = dg*16 + j;
      float dv = sD[dl], xv = sX[dl];
      float dA = __expf(dv*An[j]);
      h[j] = dA*h[j] + dv*Bv*xv;
      float part = h[j]*Cv;
      part += __shfl_xor(part, 1, 32);    // reduce the 16 states (wave32 halves)
      part += __shfl_xor(part, 2, 32);
      part += __shfl_xor(part, 4, 32);
      part += __shfl_xor(part, 8, 32);
      if(s == 0){
        int d = dbase + dl;
        float zv  = xz[r*1024 + 512 + d];
        float sil = zv/(1.f + __expf(-zv));
        ybf[r*DMQ + d] = tobf((part + dpar[d]*xv) * sil);
      }
    }
  }
}

// ---------------- fused add(+reversed add) + LayerNorm (f32 + bf16 out) ----------------
__global__ void __launch_bounds__(256)
add_ln(const float* __restrict__ a, const float* __restrict__ b2,
       const float* __restrict__ crev,
       const float* __restrict__ g, const float* __restrict__ be,
       float* __restrict__ out, __bf16* __restrict__ outbf)
{
  __shared__ float sred[8];
  int r = blockIdx.x;
  int t = threadIdx.x;
  size_t base = (size_t)r*DMQ;
  size_t rbase = 0;
  if(crev){ int b = r/LQ, l = r%LQ; rbase = ((size_t)b*LQ + (LQ-1-l))*DMQ; }
  float v0 = a[base+t]     + (b2?b2[base+t]:0.f)     + (crev?crev[rbase+t]:0.f);
  float v1 = a[base+t+256] + (b2?b2[base+t+256]:0.f) + (crev?crev[rbase+t+256]:0.f);
  float s = v0 + v1;
#pragma unroll
  for(int o = 16; o > 0; o >>= 1) s += __shfl_xor(s, o, 32);
  if((t & 31) == 0) sred[t >> 5] = s;
  __syncthreads();
  float tot = 0.f;
#pragma unroll
  for(int i = 0; i < 8; ++i) tot += sred[i];
  float m = tot * (1.f/DMQ);
  __syncthreads();
  float d0 = v0 - m, d1 = v1 - m;
  float q = d0*d0 + d1*d1;
#pragma unroll
  for(int o = 16; o > 0; o >>= 1) q += __shfl_xor(q, o, 32);
  if((t & 31) == 0) sred[t >> 5] = q;
  __syncthreads();
  float vt = 0.f;
#pragma unroll
  for(int i = 0; i < 8; ++i) vt += sred[i];
  float inv = rsqrtf(vt*(1.f/DMQ) + 1e-5f);
  float o0 = d0*inv*g[t]     + be[t];
  float o1 = d1*inv*g[t+256] + be[t+256];
  out[base+t]     = o0;
  out[base+t+256] = o1;
  if(outbf){ outbf[base+t] = tobf(o0); outbf[base+t+256] = tobf(o1); }
}

// ---------------- host orchestration ----------------
extern "C" void kernel_launch(void* const* d_in, const int* in_sizes, int n_in,
                              void* d_out, int out_size, void* d_ws, size_t ws_size,
                              hipStream_t stream)
{
  (void)in_sizes; (void)n_in; (void)out_size; (void)ws_size;
  const float* x_enc  = (const float*)d_in[0];
  const float* x_mark = (const float*)d_in[1];
  const float* emb_w  = (const float*)d_in[4];
  const float* emb_b  = (const float*)d_in[5];
  const float* inw    = (const float*)d_in[6];
  const float* cw     = (const float*)d_in[7];
  const float* cb     = (const float*)d_in[8];
  const float* xpw    = (const float*)d_in[9];
  const float* dtw    = (const float*)d_in[10];
  const float* dtb    = (const float*)d_in[11];
  const float* alog   = (const float*)d_in[12];
  const float* dpar   = (const float*)d_in[13];
  const float* ow     = (const float*)d_in[14];
  const float* w1     = (const float*)d_in[15];
  const float* b1     = (const float*)d_in[16];
  const float* w2     = (const float*)d_in[17];
  const float* b2     = (const float*)d_in[18];
  const float* ln1g   = (const float*)d_in[19];
  const float* ln1b   = (const float*)d_in[20];
  const float* ln2g   = (const float*)d_in[21];
  const float* ln2b   = (const float*)d_in[22];
  const float* fing   = (const float*)d_in[23];
  const float* finb   = (const float*)d_in[24];
  const float* gw     = (const float*)d_in[25];
  const float* gb     = (const float*)d_in[26];
  const float* pw     = (const float*)d_in[27];
  const float* pb     = (const float*)d_in[28];

  // ---- bump allocator over d_ws (64B aligned) ----
  char* base = (char*)d_ws;
  size_t off = 0;
  auto alloc = [&](size_t bytes) -> void* {
    off = (off + 63) & ~(size_t)63;
    void* p = base + off;
    off += bytes;
    return p;
  };
  const size_t RD = (size_t)RROWS*DMQ;

  // f32 buffers
  float* raw   = (float*)alloc(RD*4);
  float* enc   = (float*)alloc(RD*4);
  float* xz    = (float*)alloc(2*RD*4);            // xi cols 0..511, z cols 512..1023
  float* xc    = (float*)alloc(RD*4);
  float* dblp  = (float*)alloc((size_t)RROWS*64*4);
  float* delta = (float*)alloc(RD*4);
  float* fwdO  = (float*)alloc(RD*4);
  float* revO  = (float*)alloc(RD*4);
  float* xbuf  = (float*)alloc(RD*4);
  float* meanp = (float*)alloc((size_t)BQ*NVARQ*4);
  float* stdp  = (float*)alloc((size_t)BQ*NVARQ*4);
  // bf16 activation buffers
  __bf16* tokbf  = (__bf16*)alloc((size_t)RROWS*SEQQ*2);
  __bf16* encbf  = (__bf16*)alloc(RD*2);
  __bf16* encRbf = (__bf16*)alloc(RD*2);
  __bf16* rawbf  = (__bf16*)alloc(RD*2);
  __bf16* xcbf   = (__bf16*)alloc(RD*2);
  __bf16* dtbf   = (__bf16*)alloc((size_t)RROWS*32*2);
  __bf16* ybf    = (__bf16*)alloc(RD*2);
  __bf16* h1bf   = (__bf16*)alloc(RD*2);
  __bf16* xbufbf = (__bf16*)alloc(RD*2);
  __bf16* gatebf = (__bf16*)alloc(RD*2);
  // bf16 weight copies
  __bf16* embwbf = (__bf16*)alloc((size_t)DMQ*SEQQ*2);
  __bf16* inwbf  = (__bf16*)alloc((size_t)4*1024*512*2);
  __bf16* xpwbf  = (__bf16*)alloc((size_t)4*64*512*2);
  __bf16* dtwbf  = (__bf16*)alloc((size_t)4*512*32*2);
  __bf16* owbf   = (__bf16*)alloc((size_t)4*512*512*2);
  __bf16* w1bf   = (__bf16*)alloc((size_t)2*512*512*2);
  __bf16* w2bf   = (__bf16*)alloc((size_t)2*512*512*2);
  __bf16* gwbf   = (__bf16*)alloc((size_t)512*512*2);
  __bf16* pwbf   = (__bf16*)alloc((size_t)96*512*2);

  auto cvt = [&](const float* s, __bf16* d, int n){
    cvt_bf16<<<(n+255)/256, 256, 0, stream>>>(s, d, n);
  };
  cvt(emb_w, embwbf, DMQ*SEQQ);
  cvt(inw,   inwbf,  4*1024*512);
  cvt(xpw,   xpwbf,  4*64*512);
  cvt(dtw,   dtwbf,  4*512*32);
  cvt(ow,    owbf,   4*512*512);
  cvt(w1,    w1bf,   2*512*512);
  cvt(w2,    w2bf,   2*512*512);
  cvt(gw,    gwbf,   512*512);
  cvt(pw,    pwbf,   96*512);

  const int EW = (int)((RD + 255)/256);

  // 1) per-series normalization + bf16 token matrix (B*866, 96)
  norm_tok<<<(RROWS+255)/256, 256, 0, stream>>>(x_enc, x_mark, tokbf, meanp, stdp);
  // 2) embedding: enc = tok @ emb_w^T + emb_b   (K=96, N=512, NT=4)
  gemm_wmma<EPI_BIAS,4><<<dim3(MTILES, DMQ/64), 32, 0, stream>>>(
      tokbf, SEQQ, embwbf, SEQQ, emb_b, enc, DMQ, encbf, DMQ, DMQ);
  hipMemcpyAsync(raw,   enc,   RD*sizeof(float),  hipMemcpyDeviceToDevice, stream);
  hipMemcpyAsync(rawbf, encbf, RD*sizeof(__bf16), hipMemcpyDeviceToDevice, stream);

  for(int l = 0; l < 2; ++l){
    for(int dir = 0; dir < 2; ++dir){
      if(dir == 1)
        reverse_bf<<<EW, 256, 0, stream>>>((const unsigned short*)encbf,
                                           (unsigned short*)encRbf);
      const __bf16* src = (dir == 0) ? encbf : encRbf;
      float*       outB = (dir == 0) ? fwdO  : revO;
      size_t wo = (size_t)(l*2 + dir);
      // xz = src @ in_proj^T                       (K=512, N=1024)
      gemm_wmma<EPI_NONE,4><<<dim3(MTILES, 1024/64), 32, 0, stream>>>(
          src, DMQ, inwbf + wo*1024*512, DMQ, nullptr, xz, 1024, nullptr, 0, 0);
      // xc = silu(conv1d_k2(xi))  (f32 + bf16)
      conv_silu<<<EW, 256, 0, stream>>>(xz, cw + wo*512*2, cb + wo*512, xc, xcbf);
      // dbl = xc @ x_proj^T   (K=512, N=64: [dt|B|C]); bf16 copy of dt cols only
      gemm_wmma<EPI_NONE,4><<<dim3(MTILES, 64/64), 32, 0, stream>>>(
          xcbf, DMQ, xpwbf + wo*64*512, DMQ, nullptr, dblp, 64, dtbf, 32, 32);
      // delta = softplus(dt @ dt_w^T + dt_b)       (K=32, N=512)
      gemm_wmma<EPI_SOFTPLUS,4><<<dim3(MTILES, DMQ/64), 32, 0, stream>>>(
          dtbf, 32, dtwbf + wo*512*32, 32, dtb + wo*512, delta, DMQ, nullptr, 0, 0);
      // sequential selective scan -> ybf
      mamba_scan<<<dim3(BQ, 2), 256, 0, stream>>>(
          delta, xc, dblp, xz, alog + wo*512*16, dpar + wo*512, ybf);
      // outB = y @ out_w^T                         (K=512, N=512)
      gemm_wmma<EPI_NONE,4><<<dim3(MTILES, DMQ/64), 32, 0, stream>>>(
          ybf, DMQ, owbf + wo*512*512, DMQ, nullptr, outB, DMQ, nullptr, 0, 0);
    }
    // x = LN(enc + fwd + reverse(rev))
    add_ln<<<RROWS, 256, 0, stream>>>(enc, fwdO, revO,
                                      ln1g + l*512, ln1b + l*512, xbuf, xbufbf);
    // FFN: h1 = relu(x@w1^T+b1) -> bf16 only; y2 = h1@w2^T+b2 -> fwdO
    gemm_wmma<EPI_RELU,4><<<dim3(MTILES, DMQ/64), 32, 0, stream>>>(
        xbufbf, DMQ, w1bf + (size_t)l*512*512, DMQ, b1 + l*512,
        nullptr, 0, h1bf, DMQ, DMQ);
    gemm_wmma<EPI_BIAS,4><<<dim3(MTILES, DMQ/64), 32, 0, stream>>>(
        h1bf, DMQ, w2bf + (size_t)l*512*512, DMQ, b2 + l*512,
        fwdO, DMQ, nullptr, 0, 0);
    // enc = LN(x + y2)
    add_ln<<<RROWS, 256, 0, stream>>>(xbuf, fwdO, nullptr,
                                      ln2g + l*512, ln2b + l*512, enc, encbf);
  }
  // final LN, gated residual, fused projection + de-normalization
  add_ln<<<RROWS, 256, 0, stream>>>(enc, nullptr, nullptr, fing, finb, xbuf, xbufbf);
  gemm_gate<<<dim3(MTILES, DMQ/64), 32, 0, stream>>>(rawbf, raw, xbuf, gwbf, gb, gatebf);
  gemm_proj<<<dim3(MTILES, SEQQ/32), 32, 0, stream>>>(gatebf, pwbf, pb, meanp, stdp,
                                                      (float*)d_out);
}
